// MultiHeadAttention_77352361001314
// MI455X (gfx1250) — compile-verified, matrix operand order checked by reading, not run on hardware
//
#include <hip/hip_runtime.h>
#include <hip/hip_bf16.h>

typedef __bf16 bf16_t;
typedef __attribute__((ext_vector_type(16))) __bf16 v16bf;
typedef __attribute__((ext_vector_type(8)))  __bf16 v8bf;
typedef __attribute__((ext_vector_type(8)))  float   v8f;

static constexpr int D_MODEL   = 1024;
static constexpr int NUM_HEADS = 16;
static constexpr int D_K       = 64;
static constexpr int BATCH     = 4;
static constexpr int SEQ       = 2048;
static constexpr int M_TOT     = BATCH * SEQ;   // 8192 tokens

// ds_swizzle_b32 BITMASK mode: newlane = ((lane & and) | or) ^ xor
// offset = {xor[14:10], or[9:5], and[4:0]}; and=0x1F keeps each wave32 16-lane
// half reducing independently for xor masks < 16.
#define SWZ_XOR(v, m) \
  __int_as_float(__builtin_amdgcn_ds_swizzle(__float_as_int(v), (((m) << 10) | 0x1F)))

__device__ __forceinline__ v8f zero8() {
  v8f z = {0.f, 0.f, 0.f, 0.f, 0.f, 0.f, 0.f, 0.f};
  return z;
}

__device__ __forceinline__ v8f wmma_bf16(v16bf a, v16bf b, v8f c) {
  // (neg_a, A, neg_b, B, c_mod, C, reuse_a, reuse_b)
  return __builtin_amdgcn_wmma_f32_16x16x32_bf16(false, a, false, b, (short)0, c,
                                                 false, false);
}

// 16x32 bf16 A/B fragment from a row-major matrix (row stride ld elements).
// Lane layout (ISA 7.12.2): lanes 0-15 -> K groups {0..7,16..23},
// lanes 16-31 -> {8..15,24..31}; v16bf element i -> VGPR i/2 half i%2.
__device__ __forceinline__ v16bf load_frag(const bf16_t* __restrict__ base,
                                           int row0, int ld, int k0) {
  const int lane = threadIdx.x & 31;
  const int m  = lane & 15;
  const int kg = (lane >> 4) << 3;
  const bf16_t* p = base + (size_t)(row0 + m) * ld + (k0 + kg);
  v8bf lo = *(const v8bf*)(p);        // 16B load
  v8bf hi = *(const v8bf*)(p + 16);   // 16B load
  v16bf r;
#pragma unroll
  for (int i = 0; i < 8; ++i) { r[i] = lo[i]; r[i + 8] = hi[i]; }
  return r;
}

// Same fragment from an fp32 source, converting to bf16 in-register.
__device__ __forceinline__ v16bf load_frag(const float* __restrict__ base,
                                           int row0, int ld, int k0) {
  const int lane = threadIdx.x & 31;
  const int m  = lane & 15;
  const int kg = (lane >> 4) << 3;
  const float* p = base + (size_t)(row0 + m) * ld + (k0 + kg);
  v8f lo = *(const v8f*)(p);          // 32B (2x b128)
  v8f hi = *(const v8f*)(p + 16);
  v16bf r;
#pragma unroll
  for (int i = 0; i < 8; ++i) { r[i] = (bf16_t)lo[i]; r[i + 8] = (bf16_t)hi[i]; }
  return r;
}

// ------------------------------------------------------------- fp32 -> bf16
__global__ void cvt_f32_to_bf16_kernel(const float* __restrict__ src,
                                       bf16_t* __restrict__ dst, int n) {
  int i = blockIdx.x * blockDim.x + threadIdx.x;
  if (i < n) dst[i] = (bf16_t)src[i];
}

// ------------------------------------------------- QKV projection (bf16 WMMA)
// Y = X @ W^T (+bias); W is [out,in] so both A rows and B "columns" (= W rows)
// are contiguous along K. Wave computes a 64x64 tile; block = 4 waves = 128x128.
// transpose_v==0 -> store [b,h,s,d]; ==1 -> store V^T as [b,h,d,s].
// AT = float (convert in-register) or bf16_t (pre-converted activations).
template <typename AT>
__global__ __launch_bounds__(128)
void qkv_proj_kernel(const AT* __restrict__ X, const bf16_t* __restrict__ W,
                     const float* __restrict__ bias, bf16_t* __restrict__ out,
                     int transpose_v) {
  const int wave = threadIdx.x >> 5;
  const int lane = threadIdx.x & 31;
  const int m0 = blockIdx.y * 128 + (wave >> 1) * 64;
  const int n0 = blockIdx.x * 128 + (wave & 1) * 64;

  v8f acc[4][4];
#pragma unroll
  for (int i = 0; i < 4; ++i)
#pragma unroll
    for (int j = 0; j < 4; ++j) acc[i][j] = zero8();

  for (int kb = 0; kb < D_MODEL; kb += 32) {
    v16bf af[4], wf[4];
#pragma unroll
    for (int i = 0; i < 4; ++i) af[i] = load_frag(X, m0 + i * 16, D_MODEL, kb);
#pragma unroll
    for (int j = 0; j < 4; ++j) wf[j] = load_frag(W, n0 + j * 16, D_MODEL, kb);
#pragma unroll
    for (int i = 0; i < 4; ++i)
#pragma unroll
      for (int j = 0; j < 4; ++j)
        acc[i][j] = wmma_bf16(af[i], wf[j], acc[i][j]);
  }

  const int half = lane >> 4;
  const int nl   = lane & 15;
#pragma unroll
  for (int i = 0; i < 4; ++i)
#pragma unroll
    for (int j = 0; j < 4; ++j)
#pragma unroll
      for (int r = 0; r < 8; ++r) {
        const int tok = m0 + i * 16 + r + half * 8;
        const int col = n0 + j * 16 + nl;
        const float v = acc[i][j][r] + bias[col];
        const int b = tok >> 11, s = tok & (SEQ - 1);
        const int h = col >> 6,  d = col & (D_K - 1);
        size_t idx;
        if (transpose_v)
          idx = ((size_t)(b * NUM_HEADS + h) * D_K + d) * SEQ + s;
        else
          idx = ((size_t)(b * NUM_HEADS + h) * SEQ + s) * D_K + d;
        out[idx] = (bf16_t)v;
      }
}

// ------------------------------------------------------ flash attention core
// One wave handles 32 query rows of one (b,h); block = 4 independent waves.
// Online softmax over key tiles of 32. P goes through LDS to re-layout
// C-fragments into A-fragments for the PV WMMA.
__global__ __launch_bounds__(128)
void flash_attn_kernel(const bf16_t* __restrict__ Q, const bf16_t* __restrict__ Km,
                       const bf16_t* __restrict__ Vt, bf16_t* __restrict__ attn) {
  __shared__ __align__(16) bf16_t ldsP[4 * 32 * 32];  // 2KB per wave
  const int wave = threadIdx.x >> 5;
  const int lane = threadIdx.x & 31;
  const int bh = blockIdx.x;
  const int q0 = blockIdx.y * 128 + wave * 32;
  const int b = bh >> 4, h = bh & 15;

  const bf16_t* Qh = Q  + (size_t)bh * SEQ * D_K;
  const bf16_t* Kh = Km + (size_t)bh * SEQ * D_K;
  const bf16_t* Vh = Vt + (size_t)bh * D_K * SEQ;   // [d, s]
  bf16_t* lp = ldsP + wave * 32 * 32;

  // Q fragments for the whole 32x64 tile stay resident.
  v16bf qf[2][2];
#pragma unroll
  for (int mt = 0; mt < 2; ++mt)
#pragma unroll
    for (int kd = 0; kd < 2; ++kd)
      qf[mt][kd] = load_frag(Qh, q0 + mt * 16, D_K, kd * 32);

  v8f o[2][4];
  float mrun[2][8], lrun[2][8];
#pragma unroll
  for (int mt = 0; mt < 2; ++mt) {
#pragma unroll
    for (int nt = 0; nt < 4; ++nt) o[mt][nt] = zero8();
#pragma unroll
    for (int r = 0; r < 8; ++r) { mrun[mt][r] = -1e30f; lrun[mt][r] = 0.f; }
  }

  const float scale = 0.125f;  // 1/sqrt(64)
  const int half = lane >> 4, nl = lane & 15;

  for (int kt = 0; kt < SEQ; kt += 32) {
    // Prefetch next K/V tiles into cache (global_prefetch_b8) while this
    // iteration's softmax VALU work runs.
    if (kt + 32 < SEQ) {
      __builtin_prefetch(Kh + (size_t)(kt + 32 + lane) * D_K, 0, 1);
      __builtin_prefetch(Vh + (size_t)lane * SEQ + (kt + 32), 0, 1);
      __builtin_prefetch(Vh + (size_t)(lane + 32) * SEQ + (kt + 32), 0, 1);
    }

    // S = Q K^T for a 32x32 score tile (K rows are the "columns" of K^T).
    v16bf kf[2][2];
#pragma unroll
    for (int nk = 0; nk < 2; ++nk)
#pragma unroll
      for (int kd = 0; kd < 2; ++kd)
        kf[nk][kd] = load_frag(Kh, kt + nk * 16, D_K, kd * 32);

    v8f sc[2][2];
#pragma unroll
    for (int mt = 0; mt < 2; ++mt)
#pragma unroll
      for (int nk = 0; nk < 2; ++nk) {
        sc[mt][nk] = zero8();
#pragma unroll
        for (int kd = 0; kd < 2; ++kd)
          sc[mt][nk] = wmma_bf16(qf[mt][kd], kf[nk][kd], sc[mt][nk]);
      }

#pragma unroll
    for (int mt = 0; mt < 2; ++mt) {
      float alpha[8];
#pragma unroll
      for (int r = 0; r < 8; ++r) {
        const float s0 = sc[mt][0][r] * scale;
        const float s1 = sc[mt][1][r] * scale;
        // row max across the 16 N-lanes of this half (xor 1,2,4,8 keeps halves)
        float v = fmaxf(s0, s1);
        v = fmaxf(v, SWZ_XOR(v, 1));
        v = fmaxf(v, SWZ_XOR(v, 2));
        v = fmaxf(v, SWZ_XOR(v, 4));
        v = fmaxf(v, SWZ_XOR(v, 8));
        const float mnew = fmaxf(mrun[mt][r], v);
        alpha[r] = __expf(mrun[mt][r] - mnew);
        mrun[mt][r] = mnew;
        const float p0 = __expf(s0 - mnew);
        const float p1 = __expf(s1 - mnew);
        sc[mt][0][r] = p0; sc[mt][1][r] = p1;
        float rs = p0 + p1;
        rs += SWZ_XOR(rs, 1);
        rs += SWZ_XOR(rs, 2);
        rs += SWZ_XOR(rs, 4);
        rs += SWZ_XOR(rs, 8);
        lrun[mt][r] = lrun[mt][r] * alpha[r] + rs;
      }
#pragma unroll
      for (int nt = 0; nt < 4; ++nt)
#pragma unroll
        for (int r = 0; r < 8; ++r) o[mt][nt][r] *= alpha[r];

      // stash P tile (bf16) into this wave's LDS slice, C-layout addressing
#pragma unroll
      for (int nk = 0; nk < 2; ++nk)
#pragma unroll
        for (int r = 0; r < 8; ++r) {
          const int row = mt * 16 + r + half * 8;
          const int col = nk * 16 + nl;
          lp[row * 32 + col] = (bf16_t)sc[mt][nk][r];
        }
    }
    asm volatile("s_wait_dscnt 0" ::: "memory");  // DS stores -> DS loads

    // reload P as A-fragments (k-dim = 32 keys, one WMMA step)
    v16bf pf[2];
#pragma unroll
    for (int mt = 0; mt < 2; ++mt) {
      const int m  = lane & 15;
      const int kg = (lane >> 4) << 3;
      const bf16_t* pp = lp + (mt * 16 + m) * 32 + kg;
      v8bf lo = *(const v8bf*)pp;          // ds_load_b128
      v8bf hi = *(const v8bf*)(pp + 16);
      v16bf r;
#pragma unroll
      for (int i = 0; i < 8; ++i) { r[i] = lo[i]; r[i + 8] = hi[i]; }
      pf[mt] = r;
    }

    // O += P V ; V^T rows are contiguous along the key index.
#pragma unroll
    for (int nt = 0; nt < 4; ++nt) {
      v16bf vf = load_frag(Vh, nt * 16, SEQ, kt);
#pragma unroll
      for (int mt = 0; mt < 2; ++mt)
        o[mt][nt] = wmma_bf16(pf[mt], vf, o[mt][nt]);
    }
  }

  // normalize and store (merged heads layout [b, s, h*64+d], bf16)
#pragma unroll
  for (int mt = 0; mt < 2; ++mt) {
    float inv[8];
#pragma unroll
    for (int r = 0; r < 8; ++r) inv[r] = 1.0f / lrun[mt][r];
#pragma unroll
    for (int nt = 0; nt < 4; ++nt)
#pragma unroll
      for (int r = 0; r < 8; ++r) {
        const int srow = q0 + mt * 16 + r + half * 8;
        const int d    = nt * 16 + nl;
        attn[(size_t)(b * SEQ + srow) * D_MODEL + h * D_K + d] =
            (bf16_t)(o[mt][nt][r] * inv[r]);
      }
  }
}

// ---------------------------------------------------- output projection (f32)
__global__ __launch_bounds__(128)
void out_proj_kernel(const bf16_t* __restrict__ X, const bf16_t* __restrict__ W,
                     const float* __restrict__ bias, float* __restrict__ out) {
  const int wave = threadIdx.x >> 5;
  const int lane = threadIdx.x & 31;
  const int m0 = blockIdx.y * 128 + (wave >> 1) * 64;
  const int n0 = blockIdx.x * 128 + (wave & 1) * 64;

  v8f acc[4][4];
#pragma unroll
  for (int i = 0; i < 4; ++i)
#pragma unroll
    for (int j = 0; j < 4; ++j) acc[i][j] = zero8();

  for (int kb = 0; kb < D_MODEL; kb += 32) {
    v16bf af[4], wf[4];
#pragma unroll
    for (int i = 0; i < 4; ++i) af[i] = load_frag(X, m0 + i * 16, D_MODEL, kb);
#pragma unroll
    for (int j = 0; j < 4; ++j) wf[j] = load_frag(W, n0 + j * 16, D_MODEL, kb);
#pragma unroll
    for (int i = 0; i < 4; ++i)
#pragma unroll
      for (int j = 0; j < 4; ++j)
        acc[i][j] = wmma_bf16(af[i], wf[j], acc[i][j]);
  }

  const int half = lane >> 4;
  const int nl   = lane & 15;
#pragma unroll
  for (int i = 0; i < 4; ++i)
#pragma unroll
    for (int j = 0; j < 4; ++j)
#pragma unroll
      for (int r = 0; r < 8; ++r) {
        const int tok = m0 + i * 16 + r + half * 8;
        const int col = n0 + j * 16 + nl;
        out[(size_t)tok * D_MODEL + col] = acc[i][j][r] + bias[col];
      }
}

// ----------------------------------------------------------------- launcher
extern "C" void kernel_launch(void* const* d_in, const int* in_sizes, int n_in,
                              void* d_out, int out_size, void* d_ws, size_t ws_size,
                              hipStream_t stream) {
  (void)in_sizes; (void)n_in; (void)out_size;
  const float* query = (const float*)d_in[0];
  const float* key_  = (const float*)d_in[1];
  const float* value = (const float*)d_in[2];
  const float* w_q   = (const float*)d_in[3];
  const float* b_q   = (const float*)d_in[4];
  const float* w_k   = (const float*)d_in[5];
  const float* b_k   = (const float*)d_in[6];
  const float* w_v   = (const float*)d_in[7];
  const float* b_v   = (const float*)d_in[8];
  const float* w_o   = (const float*)d_in[9];
  const float* b_o   = (const float*)d_in[10];

  char* ws = (char*)d_ws;
  const size_t WBYTES = (size_t)D_MODEL * D_MODEL * sizeof(bf16_t);  // 2 MB
  const size_t TBYTES = (size_t)M_TOT * D_MODEL * sizeof(bf16_t);    // 16 MB
  bf16_t* wq = (bf16_t*)(ws + 0 * WBYTES);
  bf16_t* wk = (bf16_t*)(ws + 1 * WBYTES);
  bf16_t* wv = (bf16_t*)(ws + 2 * WBYTES);
  bf16_t* wo = (bf16_t*)(ws + 3 * WBYTES);
  char* t = ws + 4 * WBYTES;
  bf16_t* Qb   = (bf16_t*)(t + 0 * TBYTES);  // [b,h,s,64]
  bf16_t* Kb   = (bf16_t*)(t + 1 * TBYTES);  // [b,h,s,64]
  bf16_t* Vt   = (bf16_t*)(t + 2 * TBYTES);  // [b,h,64,s]
  bf16_t* attn = (bf16_t*)(t + 3 * TBYTES);  // [b,s,1024]
  const size_t base_bytes = 4 * WBYTES + 4 * TBYTES;          // 72 MB
  const bool have_act_ws = ws_size >= base_bytes + 3 * TBYTES; // +48 MB

  {
    const int n = D_MODEL * D_MODEL;
    dim3 cg((n + 255) / 256);
    cvt_f32_to_bf16_kernel<<<cg, 256, 0, stream>>>(w_q, wq, n);
    cvt_f32_to_bf16_kernel<<<cg, 256, 0, stream>>>(w_k, wk, n);
    cvt_f32_to_bf16_kernel<<<cg, 256, 0, stream>>>(w_v, wv, n);
    cvt_f32_to_bf16_kernel<<<cg, 256, 0, stream>>>(w_o, wo, n);
  }

  dim3 pb(128);
  dim3 pg(D_MODEL / 128, M_TOT / 128);  // (8, 64)

  if (have_act_ws) {
    // Pre-convert activations to bf16: removes per-k-step cvt VALU work and
    // halves A-side bytes in the projection inner loops.
    bf16_t* xq = (bf16_t*)(ws + base_bytes + 0 * TBYTES);
    bf16_t* xk = (bf16_t*)(ws + base_bytes + 1 * TBYTES);
    bf16_t* xv = (bf16_t*)(ws + base_bytes + 2 * TBYTES);
    const int n = M_TOT * D_MODEL;
    dim3 cg((n + 255) / 256);
    cvt_f32_to_bf16_kernel<<<cg, 256, 0, stream>>>(query, xq, n);
    cvt_f32_to_bf16_kernel<<<cg, 256, 0, stream>>>(key_,  xk, n);
    cvt_f32_to_bf16_kernel<<<cg, 256, 0, stream>>>(value, xv, n);
    qkv_proj_kernel<bf16_t><<<pg, pb, 0, stream>>>(xq, wq, b_q, Qb, 0);
    qkv_proj_kernel<bf16_t><<<pg, pb, 0, stream>>>(xk, wk, b_k, Kb, 0);
    qkv_proj_kernel<bf16_t><<<pg, pb, 0, stream>>>(xv, wv, b_v, Vt, 1);
  } else {
    qkv_proj_kernel<float><<<pg, pb, 0, stream>>>(query, wq, b_q, Qb, 0);
    qkv_proj_kernel<float><<<pg, pb, 0, stream>>>(key_,  wk, b_k, Kb, 0);
    qkv_proj_kernel<float><<<pg, pb, 0, stream>>>(value, wv, b_v, Vt, 1);
  }

  dim3 ag(BATCH * NUM_HEADS, SEQ / 128);  // (64, 16)
  flash_attn_kernel<<<ag, pb, 0, stream>>>(Qb, Kb, Vt, attn);

  out_proj_kernel<<<pg, pb, 0, stream>>>(attn, wo, b_o, (float*)d_out);
}
